// MyModel_55851754717767
// MI455X (gfx1250) — matrix operand order repaired and checked by reading, not compiled, not run on hardware
//
#include <hip/hip_runtime.h>
#include <hip/hip_bf16.h>

// ---------------------------------------------------------------------------
// Types for CDNA5 WMMA (wave32): v_wmma_f32_16x16x32_bf16
// ---------------------------------------------------------------------------
typedef __attribute__((ext_vector_type(16))) __bf16 v16bf;
typedef __attribute__((ext_vector_type(8)))  float  v8f;

union FragBF {
    v16bf        v;
    unsigned int u[8];
};

__device__ __forceinline__ unsigned short f2bf(float f) {
    unsigned int u = __float_as_uint(f);
    unsigned int r = u + 0x7FFFu + ((u >> 16) & 1u);   // round-to-nearest-even
    return (unsigned short)(r >> 16);
}
__device__ __forceinline__ float bf2f(unsigned short h) {
    return __uint_as_float(((unsigned int)h) << 16);
}

// ---------------------------------------------------------------------------
// Fragment layouts (CDNA5 ISA 7.12.2, wave32):
// A (16x32): lane m = lane&15, half = lane>>4; VGPR v -> K pair
//            kk = (v<4 ? 2v : 2v+8) + 8*half
// B (32x16): lane n = lane&15, half = lane>>4; VGPR v -> K pair
//            kk = (v<4 ? 2v : 2v+8) + 8*half
// C (16x16): lane n = lane&15; VGPR j -> row m = j + 8*(lane>>4)
// ---------------------------------------------------------------------------

__device__ __forceinline__ void load_fragA_g(FragBF& f, const unsigned short* src, int lda) {
    int lane = threadIdx.x & 31;
    int m = lane & 15, half = lane >> 4;
    const unsigned short* row = src + (size_t)m * lda;
#pragma unroll
    for (int v = 0; v < 8; ++v) {
        int kk = ((v < 4) ? (2 * v) : (2 * v + 8)) + 8 * half;
        f.u[v] = *(const unsigned int*)(row + kk);
    }
}

__device__ __forceinline__ v8f wmma_bf16(const FragBF& a, const FragBF& b, v8f c) {
    return __builtin_amdgcn_wmma_f32_16x16x32_bf16(false, a.v, false, b.v,
                                                   (short)0, c, false, false);
}

// Async global -> LDS copy of 16 bytes (CDNA5 GLOBAL_LOAD_ASYNC_TO_LDS_B128,
// tracked by ASYNCcnt). ldsDst = LDS byte address (low 32 bits of shared ptr).
__device__ __forceinline__ void async_copy_b128(void* ldsDst, const void* gSrc) {
    unsigned dst = (unsigned)(unsigned long long)ldsDst;
    unsigned long long src = (unsigned long long)gSrc;
    asm volatile("global_load_async_to_lds_b128 %0, %1, off"
                 :: "v"(dst), "v"(src)
                 : "memory");
}
__device__ __forceinline__ void async_wait0() {
    asm volatile("s_wait_asynccnt 0" ::: "memory");
}

// ---------------------------------------------------------------------------
// fp32 -> bf16 cast
// ---------------------------------------------------------------------------
__global__ void cast_bf16_kernel(const float* __restrict__ in,
                                 unsigned short* __restrict__ out, long n) {
    long i = (long)blockIdx.x * blockDim.x + threadIdx.x;
    if (i < n) out[i] = f2bf(in[i]);
}

// ---------------------------------------------------------------------------
// Tiled bf16 GEMM: C[M,N] = A[M,K] * W[N,K]^T  (A,W bf16; C bf16 or f32)
// Block: 256 thr = 8 waves (4 x 2). Block tile 128x128, wave tile 32x64, K step 32.
// ---------------------------------------------------------------------------
template <bool F32OUT>
__global__ void __launch_bounds__(256)
gemm_bf16_kernel(const unsigned short* __restrict__ A,
                 const unsigned short* __restrict__ W,
                 void* __restrict__ Cv, int M, int N, int K) {
    __shared__ unsigned short lA[128 * 32];
    __shared__ unsigned short lB[128 * 32];

    int tid   = threadIdx.x;
    int wid   = tid >> 5;
    int waveM = wid >> 1;
    int waveN = wid & 1;
    int mBase = blockIdx.y * 128;
    int nBase = blockIdx.x * 128;

    v8f c[2][4];
#pragma unroll
    for (int i = 0; i < 2; ++i)
#pragma unroll
        for (int nt = 0; nt < 4; ++nt) c[i][nt] = {};

    int srow = tid >> 1;
    int shalf = tid & 1;

    for (int k0 = 0; k0 < K; k0 += 32) {
        // async stage A/B slabs straight into LDS (no VGPR round-trip)
        async_copy_b128(lA + srow * 32 + shalf * 16,
                        A + (size_t)(mBase + srow) * K + k0 + shalf * 16);
        async_copy_b128(lA + srow * 32 + shalf * 16 + 8,
                        A + (size_t)(mBase + srow) * K + k0 + shalf * 16 + 8);
        async_copy_b128(lB + srow * 32 + shalf * 16,
                        W + (size_t)(nBase + srow) * K + k0 + shalf * 16);
        async_copy_b128(lB + srow * 32 + shalf * 16 + 8,
                        W + (size_t)(nBase + srow) * K + k0 + shalf * 16 + 8);
        if (k0 + 32 < K) {
            __builtin_prefetch(A + (size_t)(mBase + srow) * K + k0 + 32 + shalf * 16, 0, 1);
            __builtin_prefetch(W + (size_t)(nBase + srow) * K + k0 + 32 + shalf * 16, 0, 1);
        }
        async_wait0();
        __syncthreads();

        int lane = tid & 31;
        int lm = lane & 15, lhalf = lane >> 4;

        FragBF aF[2];
#pragma unroll
        for (int i = 0; i < 2; ++i) {
            int row = waveM * 32 + i * 16 + lm;
#pragma unroll
            for (int v = 0; v < 8; ++v) {
                int kk = ((v < 4) ? (2 * v) : (2 * v + 8)) + 8 * lhalf;
                aF[i].u[v] = *(const unsigned int*)&lA[row * 32 + kk];
            }
        }
        FragBF bF[4];
#pragma unroll
        for (int nt = 0; nt < 4; ++nt) {
            int row = waveN * 64 + nt * 16 + lm;
            int kb = lhalf << 4;
#pragma unroll
            for (int v = 0; v < 8; ++v)
                bF[nt].u[v] = *(const unsigned int*)&lB[row * 32 + ((v < 4) ? (2 * v) : (2 * v + 8)) + 8 * lhalf];
        }
#pragma unroll
        for (int i = 0; i < 2; ++i)
#pragma unroll
            for (int nt = 0; nt < 4; ++nt)
                c[i][nt] = wmma_bf16(aF[i], bF[nt], c[i][nt]);

        __syncthreads();
    }

    int lane = tid & 31;
    int n = lane & 15, half = lane >> 4;
#pragma unroll
    for (int i = 0; i < 2; ++i)
#pragma unroll
        for (int nt = 0; nt < 4; ++nt)
#pragma unroll
            for (int j = 0; j < 8; ++j) {
                int gm = mBase + waveM * 32 + i * 16 + j + 8 * half;
                int gn = nBase + waveN * 64 + nt * 16 + n;
                if (F32OUT)
                    ((float*)Cv)[(size_t)gm * N + gn] = c[i][nt][j];
                else
                    ((unsigned short*)Cv)[(size_t)gm * N + gn] = f2bf(c[i][nt][j]);
            }
}

// ---------------------------------------------------------------------------
// RoPE (in-place on bf16 [S, nh, 128])
// ---------------------------------------------------------------------------
__global__ void rope_kernel(unsigned short* __restrict__ X,
                            const float* __restrict__ cosp,
                            const float* __restrict__ sinp, int nh) {
    int idx = blockIdx.x * blockDim.x + threadIdx.x;
    int d = idx & 63;
    int hh = (idx >> 6) % nh;
    int s = idx / (64 * nh);
    if (s >= 2048) return;
    unsigned short* p = X + (size_t)s * nh * 128 + hh * 128;
    float x1 = bf2f(p[d]);
    float x2 = bf2f(p[d + 64]);
    float c1 = cosp[s * 128 + d], s1 = sinp[s * 128 + d];
    float c2 = cosp[s * 128 + d + 64], s2 = sinp[s * 128 + d + 64];
    p[d]      = f2bf(x1 * c1 - x2 * s1);
    p[d + 64] = f2bf(x2 * c2 + x1 * s2);
}

// ---------------------------------------------------------------------------
// Flash attention (causal, GQA 32/8, HD=128, S=2048), bf16 in/out, fp32 accum.
// Block = 128 thr = 4 waves sharing one 64-row query tile.
// K tile staged via GLOBAL_LOAD_ASYNC_TO_LDS_B128 (ASYNCcnt); V tile staged
// transposed through VGPRs; all fragment feeds are contiguous ds_load_b32/128.
// ---------------------------------------------------------------------------
#define LKS 136   // lK row stride (ushorts), 16B-aligned, bank-spreading
#define LVS 68    // lVT row stride (ushorts), bank-spreading

__global__ void __launch_bounds__(128)
attn_kernel(const unsigned short* __restrict__ Q,
            const unsigned short* __restrict__ Kg,
            const unsigned short* __restrict__ Vg,
            unsigned short* __restrict__ Ob) {
    const int DQ = 4096, DKV = 1024, HD = 128;
    const float scale = 0.08838834764831845f;  // 1/sqrt(128)

    __shared__ unsigned short lK[64 * LKS];    // K tile: [key][d]
    __shared__ unsigned short lVT[128 * LVS];  // V tile transposed: [d][key]
    __shared__ unsigned short lP[4][16 * 64];  // per-wave P scratch

    int tid  = threadIdx.x;
    int wid  = tid >> 5;
    int lane = tid & 31;
    int n    = lane & 15;
    int half = lane >> 4;
    int kb16 = half << 4;
    int h    = blockIdx.y;
    int kvh  = h >> 2;
    int qb   = blockIdx.x * 64;
    int q0   = qb + wid * 16;

    const unsigned short* Kbase = Kg + kvh * HD;
    const unsigned short* Vbase = Vg + kvh * HD;

    // Q fragments: 16 rows x 128 d
    FragBF qA[4];
#pragma unroll
    for (int kt = 0; kt < 4; ++kt)
        load_fragA_g(qA[kt], Q + (size_t)q0 * DQ + h * HD + kt * 32, DQ);

    float mrow[8], lrow[8];
    v8f acc[8];
#pragma unroll
    for (int j = 0; j < 8; ++j) { mrow[j] = -3.0e38f; lrow[j] = 0.0f; }
#pragma unroll
    for (int dt = 0; dt < 8; ++dt) acc[dt] = {};

    int srow = tid >> 1;   // staging: key row 0..63
    int sh   = tid & 1;    // staging: 64-wide d half

    for (int j0 = 0; j0 < qb + 64; j0 += 64) {
        // ---- K tile: async straight to LDS (row-major, padded stride) ----
#pragma unroll
        for (int i = 0; i < 8; ++i)
            async_copy_b128(lK + srow * LKS + sh * 64 + i * 8,
                            Kbase + (size_t)(j0 + srow) * DKV + sh * 64 + i * 8);

        // ---- V tile: load through VGPRs, store transposed [d][key] ----
        {
            const uint4* gV = (const uint4*)(Vbase + (size_t)(j0 + srow) * DKV + sh * 64);
#pragma unroll
            for (int i = 0; i < 8; ++i) {
                uint4 vv = gV[i];
                const unsigned short* e = (const unsigned short*)&vv;
#pragma unroll
                for (int ee = 0; ee < 8; ++ee) {
                    int d = sh * 64 + i * 8 + ee;
                    lVT[d * LVS + srow] = e[ee];
                }
            }
        }
        if (j0 + 64 < qb + 64) {   // prefetch next V tile (global_prefetch_b8)
            __builtin_prefetch(Vbase + (size_t)(j0 + 64 + srow) * DKV + sh * 64, 0, 1);
        }
        async_wait0();
        __syncthreads();

        // ---- scores: S = Q * K^T (16 x 64), B[k=d][n=key] = lK[key][d] ----
        v8f sc[4];
#pragma unroll
        for (int nt = 0; nt < 4; ++nt) {
            sc[nt] = {};
            FragBF bK;
#pragma unroll
            for (int kt = 0; kt < 4; ++kt) {
#pragma unroll
                for (int v = 0; v < 8; ++v)
                    bK.u[v] = *(const unsigned int*)&lK[(nt * 16 + n) * LKS + kt * 32 + ((v < 4) ? (2 * v) : (2 * v + 8)) + 8 * half];
                sc[nt] = wmma_bf16(qA[kt], bK, sc[nt]);
            }
        }

        // ---- scale + causal mask + row max ----
        float newm[8];
#pragma unroll
        for (int j = 0; j < 8; ++j) {
            int qr = q0 + j + 8 * half;
            float mx = -3.0e38f;
#pragma unroll
            for (int nt = 0; nt < 4; ++nt) {
                float v = sc[nt][j] * scale;
                int key = j0 + nt * 16 + n;
                if (key > qr) v = -1.0e9f;
                sc[nt][j] = v;
                mx = fmaxf(mx, v);
            }
            mx = fmaxf(mx, __shfl_xor(mx, 1, 32));
            mx = fmaxf(mx, __shfl_xor(mx, 2, 32));
            mx = fmaxf(mx, __shfl_xor(mx, 4, 32));
            mx = fmaxf(mx, __shfl_xor(mx, 8, 32));
            newm[j] = fmaxf(mrow[j], mx);
        }

        // ---- P = exp(S - newm); row sums; spill P (row-major) to LDS ----
        float rs[8];
#pragma unroll
        for (int j = 0; j < 8; ++j) rs[j] = 0.0f;
#pragma unroll
        for (int nt = 0; nt < 4; ++nt)
#pragma unroll
            for (int j = 0; j < 8; ++j) {
                float p = __expf(sc[nt][j] - newm[j]);
                rs[j] += p;
                lP[wid][(j + 8 * half) * 64 + nt * 16 + n] = f2bf(p);
            }
#pragma unroll
        for (int j = 0; j < 8; ++j) {
            rs[j] += __shfl_xor(rs[j], 1, 32);
            rs[j] += __shfl_xor(rs[j], 2, 32);
            rs[j] += __shfl_xor(rs[j], 4, 32);
            rs[j] += __shfl_xor(rs[j], 8, 32);
            float al = __expf(mrow[j] - newm[j]);
            lrow[j] = lrow[j] * al + rs[j];
            mrow[j] = newm[j];
#pragma unroll
            for (int dt = 0; dt < 8; ++dt) acc[dt][j] *= al;
        }

        // ---- reload P in A-layout (2 K-slabs of 32 keys) ----
        FragBF pA[2];
#pragma unroll
        for (int ks = 0; ks < 2; ++ks)
#pragma unroll
            for (int v = 0; v < 8; ++v) {
                int kk = ((v < 4) ? (2 * v) : (2 * v + 8)) + 8 * half;
                pA[ks].u[v] = *(const unsigned int*)&lP[wid][n * 64 + ks * 32 + kk];
            }

        // ---- acc += P * V, B[k=key][n=d] = lVT[d][key] ----
#pragma unroll
        for (int dt = 0; dt < 8; ++dt)
#pragma unroll
            for (int ks = 0; ks < 2; ++ks) {
                FragBF bV;
#pragma unroll
                for (int v = 0; v < 8; ++v)
                    bV.u[v] = *(const unsigned int*)&lVT[(dt * 16 + n) * LVS + ks * 32 + ((v < 4) ? (2 * v) : (2 * v + 8)) + 8 * half];
                acc[dt] = wmma_bf16(pA[ks], bV, acc[dt]);
            }
        __syncthreads();
    }

    // ---- epilogue: normalize, store bf16 [S, 32*128] ----
#pragma unroll
    for (int dt = 0; dt < 8; ++dt)
#pragma unroll
        for (int j = 0; j < 8; ++j) {
            int sr = q0 + j + 8 * half;
            float o = acc[dt][j] / lrow[j];
            Ob[(size_t)sr * DQ + h * HD + dt * 16 + n] = f2bf(o);
        }
}

// ---------------------------------------------------------------------------
// Host launch
// ---------------------------------------------------------------------------
extern "C" void kernel_launch(void* const* d_in, const int* in_sizes, int n_in,
                              void* d_out, int out_size, void* d_ws, size_t ws_size,
                              hipStream_t stream) {
    const float* hs   = (const float*)d_in[0];  // [1,2048,4096]
    const float* cosp = (const float*)d_in[2];  // [1,2048,128]
    const float* sinp = (const float*)d_in[3];
    const float* Wq   = (const float*)d_in[6];  // [4096,4096]
    const float* Wk   = (const float*)d_in[7];  // [1024,4096]
    const float* Wv   = (const float*)d_in[8];  // [1024,4096]
    const float* Wo   = (const float*)d_in[9];  // [4096,4096]

    char* ws = (char*)d_ws;
    const size_t MB = 1u << 20;
    unsigned short* Xb  = (unsigned short*)(ws + 0);         // 16 MB
    unsigned short* Wqb = (unsigned short*)(ws + 16 * MB);   // 32 MB
    unsigned short* Wkb = (unsigned short*)(ws + 48 * MB);   // 8 MB
    unsigned short* Wvb = (unsigned short*)(ws + 56 * MB);   // 8 MB
    unsigned short* Wob = (unsigned short*)(ws + 64 * MB);   // 32 MB
    unsigned short* Qb  = (unsigned short*)(ws + 96 * MB);   // 16 MB
    unsigned short* Kbf = (unsigned short*)(ws + 112 * MB);  // 4 MB
    unsigned short* Vbf = (unsigned short*)(ws + 116 * MB);  // 4 MB
    unsigned short* Ab  = (unsigned short*)(ws + 120 * MB);  // 16 MB

    const long nX = 2048L * 4096, nWq = 4096L * 4096, nWkv = 1024L * 4096;

    cast_bf16_kernel<<<(int)(nX  / 256), 256, 0, stream>>>(hs, Xb, nX);
    cast_bf16_kernel<<<(int)(nWq / 256), 256, 0, stream>>>(Wq, Wqb, nWq);
    cast_bf16_kernel<<<(int)(nWkv/ 256), 256, 0, stream>>>(Wk, Wkb, nWkv);
    cast_bf16_kernel<<<(int)(nWkv/ 256), 256, 0, stream>>>(Wv, Wvb, nWkv);
    cast_bf16_kernel<<<(int)(nWq / 256), 256, 0, stream>>>(Wo, Wob, nWq);

    // Q/K/V projections (bf16 out)
    gemm_bf16_kernel<false><<<dim3(4096 / 128, 2048 / 128), 256, 0, stream>>>(Xb, Wqb, Qb, 2048, 4096, 4096);
    gemm_bf16_kernel<false><<<dim3(1024 / 128, 2048 / 128), 256, 0, stream>>>(Xb, Wkb, Kbf, 2048, 1024, 4096);
    gemm_bf16_kernel<false><<<dim3(1024 / 128, 2048 / 128), 256, 0, stream>>>(Xb, Wvb, Vbf, 2048, 1024, 4096);

    // RoPE on Q (32 heads) and K (8 heads)
    rope_kernel<<<(2048 * 32 * 64) / 256, 256, 0, stream>>>(Qb, cosp, sinp, 32);
    rope_kernel<<<(2048 * 8 * 64) / 256, 256, 0, stream>>>(Kbf, cosp, sinp, 8);

    // Flash attention: grid (S/64 q-tiles, 32 heads), 4 waves/block
    attn_kernel<<<dim3(2048 / 64, 32), 128, 0, stream>>>(Qb, Kbf, Vbf, Ab);

    // Output projection (fp32 out)
    gemm_bf16_kernel<true><<<dim3(4096 / 128, 2048 / 128), 256, 0, stream>>>(Ab, Wob, (float*)d_out, 2048, 4096, 4096);
}
